// NanoCTM_51041391346316
// MI455X (gfx1250) — matrix operand order repaired and checked by reading, not compiled, not run on hardware
//
#include <hip/hip_runtime.h>
#include <hip/hip_bf16.h>

typedef __bf16 bf16;
typedef __attribute__((ext_vector_type(16))) __bf16 v16bf;
typedef __attribute__((ext_vector_type(8)))  __bf16 v8bf;
typedef __attribute__((ext_vector_type(8)))  float  v8f;

namespace {
constexpr int kB   = 128;
constexpr int kS   = 2048;
constexpr int kDin = 512;
constexpr int kDm  = 2048;
constexpr int kM   = 25;
constexpr int kH   = 8;
constexpr int kT   = 10;
constexpr int kRep = 2080;   // 64*65/2
constexpr int kOut = 4096;   // S*2
constexpr float kPi = 3.14159265358979323846f;
constexpr float kInvLn2 = 1.4426950408889634f;
}

// ---------------------------------------------------------------------------
// Generic bf16 WMMA GEMM: C[M,N] = A[M,K] @ Bt[N,K]^T (+bias), batched over z.
// A row-major bf16 (lda), Bt is B transposed, row-major bf16 (ldb).
// One wave computes one 16x16 C tile; block = 4 waves along N.
// Requires M%16==0, N%64==0, K%32==0 (true for all call sites here).
// Fragment layouts follow CDNA5 ISA 7.12.2 (16-bit A 16x32 / B 32x16).
// ---------------------------------------------------------------------------
__global__ __launch_bounds__(128) void gemm_bf16(
    const bf16* __restrict__ A, const bf16* __restrict__ Bt,
    const float* __restrict__ bias, void* __restrict__ C,
    int M, int N, int K, int lda, int ldb, int ldc,
    long sAz, long sBz, long sCz, int c_is_bf16)
{
  const int lane = threadIdx.x & 31;
  const int wv   = threadIdx.x >> 5;
  const int tn   = (blockIdx.x * 4 + wv) * 16;
  const int tm   = blockIdx.y * 16;
  const int z    = blockIdx.z;
  if (tn >= N || tm >= M) return;
  A  += (long)z * sAz;
  Bt += (long)z * sBz;
  const int  l15 = lane & 15;
  const bool hi  = lane >= 16;
  const bf16* arow = A  + (long)(tm + l15) * lda;
  const bf16* brow = Bt + (long)(tn + l15) * ldb;
  v8f acc = {};
  for (int k0 = 0; k0 < K; k0 += 32) {
    // A 16x32: lanes 0-15 -> K {k0..k0+7, k0+16..k0+23}; lanes 16-31 -> {+8..+15, +24..+31}
    const bf16* ap = arow + k0 + (hi ? 8 : 0);
    v8bf a0 = *(const v8bf*)(ap);
    v8bf a1 = *(const v8bf*)(ap + 16);
    // B 32x16 (as Bt rows): lanes 0-15 -> K {k0..k0+7, k0+16..k0+23}; lanes 16-31 -> {+8..+15, +24..+31}
    const bf16* bp = brow + k0 + (hi ? 8 : 0);
    v8bf b0 = *(const v8bf*)(bp);
    v8bf b1 = *(const v8bf*)(bp + 16);
    v16bf av, bvv;
#pragma unroll
    for (int i = 0; i < 8; ++i) {
      av[i] = a0[i]; av[i + 8] = a1[i];
      bvv[i] = b0[i]; bvv[i + 8] = b1[i];
    }
    acc = __builtin_amdgcn_wmma_f32_16x16x32_bf16(
        false, av, false, bvv, (short)0, acc, false, false);
  }
  const int col = tn + l15;
  const float bval = bias ? bias[col] : 0.0f;
  const int rbase = tm + (hi ? 8 : 0);
  if (c_is_bf16) {
    bf16* Cb = (bf16*)C + (long)z * sCz;
#pragma unroll
    for (int r = 0; r < 8; ++r)
      Cb[(long)(rbase + r) * ldc + col] = (bf16)(acc[r] + bval);
  } else {
    float* Cf = (float*)C + (long)z * sCz;
#pragma unroll
    for (int r = 0; r < 8; ++r)
      Cf[(long)(rbase + r) * ldc + col] = acc[r] + bval;
  }
}

// out[c*R + r] = bf16(in[r*C + c])   (transpose + downcast)
__global__ void tcast_f32_bf16(const float* __restrict__ in, bf16* __restrict__ out,
                               int R, int C)
{
  long i = (long)blockIdx.x * 256 + threadIdx.x;
  long total = (long)R * C;
  if (i >= total) return;
  int r = (int)(i / C), c = (int)(i % C);
  out[(long)c * R + r] = (bf16)in[i];
}

// F[t*S+s, d] = bf16(emb[t,d] + pe[s,d]),  pe = -sin*Wp0 + cos*Wp1 + bpos
__global__ void build_F(const float* __restrict__ emb, const float* __restrict__ W_pos,
                        const float* __restrict__ b_pos, bf16* __restrict__ F)
{
  long i = (long)blockIdx.x * 256 + threadIdx.x;
  if (i >= (long)2 * kS * kDin) return;
  int d = (int)(i & (kDin - 1));
  int f = (int)(i >> 9);
  int t = f >> 11;
  int s = f & (kS - 1);
  float th = kPi * (float)s / (float)(kS - 1);
  float pe = -__sinf(th) * W_pos[d] + __cosf(th) * W_pos[kDin + d] + b_pos[d];
  F[i] = (bf16)(emb[t * kDin + d] + pe);
}

__global__ void init_iuju(int* __restrict__ iu, int* __restrict__ ju)
{
  if (threadIdx.x == 0 && blockIdx.x == 0) {
    int p = 0;
    for (int i = 0; i < 64; ++i)
      for (int j = i; j < 64; ++j) { iu[p] = i; ju[p] = j; ++p; }
  }
}

__global__ void bcast_act(const float* __restrict__ sa, float* __restrict__ act)
{
  long i = (long)blockIdx.x * 256 + threadIdx.x;
  if (i < (long)kB * kDm) act[i] = sa[i % kDm];
}

__global__ void bcast_trace(const float* __restrict__ st, float* __restrict__ trace)
{
  long i = (long)blockIdx.x * 256 + threadIdx.x;
  if (i < (long)kB * kDm * kM) trace[i] = st[i % ((long)kDm * kM)];
}

__device__ __forceinline__ float wred_max(float v) {
  for (int off = 16; off; off >>= 1) v = fmaxf(v, __shfl_xor(v, off, 32));
  return v;
}
__device__ __forceinline__ float wred_sum(float v) {
  for (int off = 16; off; off >>= 1) v += __shfl_xor(v, off, 32);
  return v;
}

// sync module: pair = sel[iu]*sel[ju]; alpha,beta EMA; s = alpha/sqrt(beta) (bf16)
__global__ __launch_bounds__(256) void sync_step(
    const float* __restrict__ act, float* __restrict__ alpha, float* __restrict__ beta,
    const float* __restrict__ decay, const int* __restrict__ iu, const int* __restrict__ ju,
    bf16* __restrict__ sout, int offset, int first)
{
  int b = blockIdx.x, tid = threadIdx.x;
  __shared__ float sel[64];
  if (tid < 64) sel[tid] = act[(long)b * kDm + offset + tid];
  __syncthreads();
  for (int p = tid; p < kRep; p += 256) {
    float pr = sel[iu[p]] * sel[ju[p]];
    long idx = (long)b * kRep + p;
    float a, bb;
    if (first) { a = pr; bb = 1.0f; }
    else {
      float r = __expf(-decay[p]);
      a = r * alpha[idx] + pr;
      bb = r * beta[idx] + 1.0f;
    }
    alpha[idx] = a;
    beta[idx]  = bb;
    sout[idx]  = (bf16)(a * rsqrtf(bb));
  }
}

// Per (h,b): select score by token, softmax over S, write masked probs (bf16)
// scores layout [tok][h][b][s]; attn layout [h][b][tok*S + s]
__global__ __launch_bounds__(256) void softmax_select(
    const float* __restrict__ scores, const int* __restrict__ tokens,
    bf16* __restrict__ attn)
{
  int h = blockIdx.x >> 7, b = blockIdx.x & 127;
  const float* s0 = scores + ((long)h * kB + b) * kS;
  const float* s1 = scores + ((long)(kH + h) * kB + b) * kS;
  const int* tk = tokens + (long)b * kS;
  __shared__ float sred[8];
  __shared__ float s_mx, s_inv;
  int tid = threadIdx.x, lane = tid & 31, wid = tid >> 5;
  float mx = -3.4e38f;
  for (int s = tid; s < kS; s += 256) {
    float v = (tk[s] ? s1[s] : s0[s]) * 0.125f;
    mx = fmaxf(mx, v);
  }
  mx = wred_max(mx);
  if (lane == 0) sred[wid] = mx;
  __syncthreads();
  if (tid == 0) {
    float m = sred[0];
    for (int i = 1; i < 8; ++i) m = fmaxf(m, sred[i]);
    s_mx = m;
  }
  __syncthreads();
  float M0 = s_mx, sum = 0.0f;
  for (int s = tid; s < kS; s += 256) {
    float v = (tk[s] ? s1[s] : s0[s]) * 0.125f;
    sum += __expf(v - M0);
  }
  sum = wred_sum(sum);
  if (lane == 0) sred[wid] = sum;
  __syncthreads();
  if (tid == 0) {
    float zt = 0;
    for (int i = 0; i < 8; ++i) zt += sred[i];
    s_inv = 1.0f / zt;
  }
  __syncthreads();
  float inv = s_inv;
  bf16* a = attn + ((long)h * kB + b) * 2 * kS;
  for (int s = tid; s < kS; s += 256) {
    int tok = tk[s];
    float v = (tok ? s1[s] : s0[s]) * 0.125f;
    float p = __expf(v - M0) * inv;
    a[s]      = tok ? (bf16)0.0f : (bf16)p;
    a[kS + s] = tok ? (bf16)p : (bf16)0.0f;
  }
}

__global__ void pack_act(const float* __restrict__ act, bf16* __restrict__ cat)
{
  long i = (long)blockIdx.x * 256 + threadIdx.x;
  if (i >= (long)kB * kDm) return;
  int b = (int)(i >> 11), n = (int)(i & (kDm - 1));
  cat[(long)b * 2560 + 512 + n] = (bf16)act[i];
}

// state = LN(g1*sigmoid(g2)); circular trace write; per-neuron LN over 25; NLM -> act
__global__ __launch_bounds__(256) void state_update(
    const float* __restrict__ pre, float* __restrict__ trace,
    const float* __restrict__ ln_s_g, const float* __restrict__ ln_s_b,
    const float* __restrict__ ln_n_g, const float* __restrict__ ln_n_b,
    const float* __restrict__ Wn, const float* __restrict__ bn,
    const float* __restrict__ temp, float* __restrict__ act, int t)
{
  int b = blockIdx.x, tid = threadIdx.x, lane = tid & 31, wid = tid >> 5;
  __shared__ float hbuf[kDm];
  __shared__ float r1[8], r2[8];
  __shared__ float s_mu, s_rstd;
  const float* pb = pre + (long)b * 4096;
  float a1 = 0.0f, a2 = 0.0f;
  for (int n = tid; n < kDm; n += 256) {
    float g1 = pb[n], g2 = pb[kDm + n];
    float hv = g1 * (1.0f / (1.0f + __expf(-g2)));
    hbuf[n] = hv;
    a1 += hv; a2 += hv * hv;
  }
  a1 = wred_sum(a1); a2 = wred_sum(a2);
  if (lane == 0) { r1[wid] = a1; r2[wid] = a2; }
  __syncthreads();
  if (tid == 0) {
    float s = 0, q = 0;
    for (int i = 0; i < 8; ++i) { s += r1[i]; q += r2[i]; }
    float mu = s / (float)kDm;
    s_mu = mu;
    s_rstd = rsqrtf(q / (float)kDm - mu * mu + 1e-5f);
  }
  __syncthreads();
  float mu = s_mu, rstd = s_rstd, invT = 1.0f / temp[0];
  float* trb = trace + (long)b * kDm * kM;
  int w = t % kM;
  for (int n = tid; n < kDm; n += 256) {
    float st = (hbuf[n] - mu) * rstd * ln_s_g[n] + ln_s_b[n];
    float* tr = trb + (long)n * kM;
    tr[w] = st;
    float tv[kM];
    float m1 = 0.0f, m2 = 0.0f;
#pragma unroll
    for (int m = 0; m < kM; ++m) { float v = tr[m]; tv[m] = v; m1 += v; m2 += v * v; }
    m1 *= (1.0f / kM);
    m2 = m2 * (1.0f / kM) - m1 * m1;
    float rs = rsqrtf(m2 + 1e-5f);
    float y0 = bn[n * 2 + 0], y1 = bn[n * 2 + 1];
#pragma unroll
    for (int m = 0; m < kM; ++m) {
      int j = (t + 1 + m) % kM;           // logical index m <-> physical slot j
      float x = (tv[j] - m1) * rs * ln_n_g[m] + ln_n_b[m];
      y0 += x * Wn[(m * 2 + 0) * kDm + n];
      y1 += x * Wn[(m * 2 + 1) * kDm + n];
    }
    act[(long)b * kDm + n] = y0 * (1.0f / (1.0f + __expf(-y1))) * invT;
  }
}

// scatter preds into stacked (B, 4096, T) layout; compute entropy certs
__global__ __launch_bounds__(256) void finalize_step(
    const float* __restrict__ predbuf, float* __restrict__ out, int t)
{
  int b = blockIdx.x, tid = threadIdx.x, lane = tid & 31, wid = tid >> 5;
  __shared__ float sred[8];
  const float* pb = predbuf + (long)b * kOut;
  float* ob = out + (long)b * kOut * kT;
  for (int o = tid; o < kOut; o += 256) ob[o * kT + t] = pb[o];
  float se = 0.0f;
  for (int s = tid; s < kS; s += 256) {
    float l0 = pb[2 * s], l1 = pb[2 * s + 1];
    float m = fmaxf(l0, l1);
    float e0 = __expf(l0 - m), e1 = __expf(l1 - m);
    float z = e0 + e1;
    float ent = (__logf(z) - (e0 * (l0 - m) + e1 * (l1 - m)) / z) * kInvLn2;
    se += ent;
  }
  se = wred_sum(se);
  if (lane == 0) sred[wid] = se;
  __syncthreads();
  if (tid == 0) {
    float tot = 0;
    for (int i = 0; i < 8; ++i) tot += sred[i];
    float ne = tot / (float)kS;
    float* cb = out + (long)kB * kOut * kT + (long)b * 2 * kT;
    cb[t] = ne;
    cb[kT + t] = 1.0f - ne;
  }
}

// ---------------------------------------------------------------------------
extern "C" void kernel_launch(void* const* d_in, const int* in_sizes, int n_in,
                              void* d_out, int out_size, void* d_ws, size_t ws_size,
                              hipStream_t stream)
{
  (void)in_sizes; (void)n_in; (void)out_size; (void)ws_size;
  const int*   tokens      = (const int*)d_in[0];
  const float* emb         = (const float*)d_in[1];
  const float* W_pos       = (const float*)d_in[2];
  const float* b_pos       = (const float*)d_in[3];
  const float* Wq          = (const float*)d_in[4];
  const float* bq          = (const float*)d_in[5];
  const float* Wk          = (const float*)d_in[6];
  const float* bk          = (const float*)d_in[7];
  const float* Wv          = (const float*)d_in[8];
  const float* bv          = (const float*)d_in[9];
  const float* Wo          = (const float*)d_in[10];
  const float* bo          = (const float*)d_in[11];
  const float* Ws          = (const float*)d_in[12];
  const float* bs          = (const float*)d_in[13];
  const float* ln_s_g      = (const float*)d_in[14];
  const float* ln_s_b      = (const float*)d_in[15];
  const float* ln_n_g      = (const float*)d_in[16];
  const float* ln_n_b      = (const float*)d_in[17];
  const float* Wn          = (const float*)d_in[18];
  const float* bn          = (const float*)d_in[19];
  const float* temp        = (const float*)d_in[20];
  const float* Wout        = (const float*)d_in[21];
  const float* bout        = (const float*)d_in[22];
  const float* start_act   = (const float*)d_in[23];
  const float* start_trace = (const float*)d_in[24];
  const float* dec_a       = (const float*)d_in[25];
  const float* dec_o       = (const float*)d_in[26];
  float* out = (float*)d_out;

  char* base = (char*)d_ws;
  size_t off = 0;
  auto alloc = [&](size_t bytes) -> char* {
    char* p = base + off;
    off += (bytes + 255) & ~(size_t)255;
    return p;
  };
  int*  iu    = (int*)alloc((size_t)kRep * 4);
  int*  ju    = (int*)alloc((size_t)kRep * 4);
  bf16* WqT   = (bf16*)alloc((size_t)kDin * kRep * 2);          // (512, 2080)
  bf16* WkT   = (bf16*)alloc((size_t)kDin * kDin * 2);          // (512, 512)
  bf16* WvT   = (bf16*)alloc((size_t)kDin * kDin * 2);
  bf16* WoT   = (bf16*)alloc((size_t)kDin * kDin * 2);
  bf16* WsT   = (bf16*)alloc((size_t)4096 * 2560 * 2);          // (4096, 2560)
  bf16* WoutT = (bf16*)alloc((size_t)kOut * kRep * 2);          // (4096, 2080)
  bf16* F     = (bf16*)alloc((size_t)2 * kS * kDin * 2);        // (4096, 512)
  bf16* Kbf   = (bf16*)alloc((size_t)2 * kS * kDin * 2);        // (4096, 512)
  bf16* Vt    = (bf16*)alloc((size_t)kDin * 2 * kS * 2);        // (512, 4096)
  float* act  = (float*)alloc((size_t)kB * kDm * 4);
  float* trace= (float*)alloc((size_t)kB * kDm * kM * 4);
  float* a_a  = (float*)alloc((size_t)kB * kRep * 4);
  float* b_a  = (float*)alloc((size_t)kB * kRep * 4);
  float* a_o  = (float*)alloc((size_t)kB * kRep * 4);
  float* b_o  = (float*)alloc((size_t)kB * kRep * 4);
  bf16* s_act = (bf16*)alloc((size_t)kB * kRep * 2);
  bf16* s_out = (bf16*)alloc((size_t)kB * kRep * 2);
  bf16* Qb    = (bf16*)alloc((size_t)kB * kDin * 2);
  float* scores = (float*)alloc((size_t)2 * kH * kB * kS * 4);  // also V f32 temp
  bf16* attnb = (bf16*)alloc((size_t)kH * kB * 2 * kS * 2);
  bf16* obuf  = (bf16*)alloc((size_t)kB * kDin * 2);
  bf16* catb  = (bf16*)alloc((size_t)kB * 2560 * 2);
  float* pre  = (float*)alloc((size_t)kB * 4096 * 4);
  float* predb= (float*)alloc((size_t)kB * kOut * 4);

  auto gemm = [&](const bf16* A, const bf16* Bt, const float* bias, void* C,
                  int M, int N, int K, int lda, int ldb, int ldc,
                  long sAz, long sBz, long sCz, int nz, int cbf) {
    dim3 g(N / 64, M / 16, nz);
    gemm_bf16<<<g, 128, 0, stream>>>(A, Bt, bias, C, M, N, K, lda, ldb, ldc,
                                     sAz, sBz, sCz, cbf);
  };
  auto tcast = [&](const float* in, bf16* o, int R, int C) {
    long tot = (long)R * C;
    tcast_f32_bf16<<<(unsigned)((tot + 255) / 256), 256, 0, stream>>>(in, o, R, C);
  };

  // ---- one-time (per call) precompute ----
  init_iuju<<<1, 32, 0, stream>>>(iu, ju);
  tcast(Wq,   WqT,   kRep, kDin);
  tcast(Wk,   WkT,   kDin, kDin);
  tcast(Wv,   WvT,   kDin, kDin);
  tcast(Wo,   WoT,   kDin, kDin);
  tcast(Ws,   WsT,   2560, 4096);
  tcast(Wout, WoutT, kRep, kOut);
  build_F<<<(2 * kS * kDin) / 256, 256, 0, stream>>>(emb, W_pos, b_pos, F);
  // K = F @ Wk + bk  (bf16, row-major, rows = tok*S + s)
  gemm(F, WkT, bk, Kbf, 2 * kS, kDin, kDin, kDin, kDin, kDin, 0, 0, 0, 1, 1);
  // V = F @ Wv + bv  (f32 temp in scores buffer), then transpose -> Vt (512, 4096)
  gemm(F, WvT, bv, scores, 2 * kS, kDin, kDin, kDin, kDin, kDin, 0, 0, 0, 1, 0);
  tcast(scores, Vt, 2 * kS, kDin);
  bcast_act<<<(kB * kDm) / 256, 256, 0, stream>>>(start_act, act);
  bcast_trace<<<(kB * kDm * kM) / 256, 256, 0, stream>>>(start_trace, trace);

  // ---- T internal "thought" iterations ----
  for (int t = 0; t < kT; ++t) {
    // action sync: last 64 neurons of act
    sync_step<<<kB, 256, 0, stream>>>(act, a_a, b_a, dec_a, iu, ju, s_act,
                                      kDm - 64, t == 0 ? 1 : 0);
    // q = s_act @ Wq + bq  (bf16)
    gemm(s_act, WqT, bq, Qb, kB, kDin, kRep, kRep, kRep, kDin, 0, 0, 0, 1, 1);
    // scores[tok][h] = Q_h @ K_{tok,h}^T  (batched over heads)
    gemm(Qb, Kbf, nullptr, scores,
         kB, kS, 64, kDin, kDin, kS, 64, 64, (long)kB * kS, kH, 0);
    gemm(Qb, Kbf + (long)kS * kDin, nullptr, scores + (long)kH * kB * kS,
         kB, kS, 64, kDin, kDin, kS, 64, 64, (long)kB * kS, kH, 0);
    // token-select + softmax + mask-split -> attn (bf16)
    softmax_select<<<kH * kB, 256, 0, stream>>>(scores, tokens, attnb);
    // o_h = [attn0|attn1]_h @ [V0;V1]_h   (bf16 into obuf[b, h*64+d])
    gemm(attnb, Vt, nullptr, obuf,
         kB, 64, 2 * kS, 2 * kS, 2 * kS, kDin,
         (long)kB * 2 * kS, (long)64 * 2 * kS, 64, kH, 1);
    // o @ Wo + bo -> cat[:, 0:512] (bf16, ldc = 2560)
    gemm(obuf, WoT, bo, catb, kB, kDin, kDin, kDin, kDin, 2560, 0, 0, 0, 1, 1);
    // cat[:, 512:2560] = act (uses OLD act, before state update)
    pack_act<<<(kB * kDm) / 256, 256, 0, stream>>>(act, catb);
    // pre = cat @ Ws + bs (f32)
    gemm(catb, WsT, bs, pre, kB, 4096, 2560, 2560, 2560, 4096, 0, 0, 0, 1, 0);
    // gated LN state -> circular trace -> per-neuron LN -> NLM -> new act
    state_update<<<kB, 256, 0, stream>>>(pre, trace, ln_s_g, ln_s_b,
                                         ln_n_g, ln_n_b, Wn, bn, temp, act, t);
    // output sync: first 64 neurons of NEW act
    sync_step<<<kB, 256, 0, stream>>>(act, a_o, b_o, dec_o, iu, ju, s_out,
                                      0, t == 0 ? 1 : 0);
    // pred = s_out @ Wout + bout (f32)
    gemm(s_out, WoutT, bout, predb, kB, kOut, kRep, kRep, kRep, kOut, 0, 0, 0, 1, 0);
    // scatter into (B, 4096, T) + entropy certs
    finalize_step<<<kB, 256, 0, stream>>>(predb, out, t);
  }
}
